// TipAdapterHead_28948079575487
// MI455X (gfx1250) — matrix operand order repaired and checked by reading, not compiled, not run on hardware
//
#include <hip/hip_runtime.h>
#include <math.h>

// ---------------------------------------------------------------------------
// Tip-Adapter head, MI455X (gfx1250), wave32 + v_wmma_f32_16x16x32_f16.
//   out = q + ALPHA * exp(-BETA*(1 - l2n(q) @ l2n(K)^T)) @ l2n(K)
// Fused two-GEMM kernel; affinity tile lives only in LDS (f16, double-buffered).
// ---------------------------------------------------------------------------

#define NQ    4096
#define NK    16384
#define DM    512
#define BETA_  5.5f
#define ALPHA_ 0.5f

typedef _Float16 h16;
typedef __attribute__((ext_vector_type(16))) _Float16 v16h;
typedef __attribute__((ext_vector_type(8)))  _Float16 v8h;
typedef __attribute__((ext_vector_type(8)))  float    v8f;
typedef __attribute__((ext_vector_type(4)))  float    v4f;

union V16 { v16h v; v8h h[2]; };

// A-matrix 16x32 f16 fragment: per lane, 16 contiguous K values
// (lanes 0-15: K=0..15, lanes 16-31: K=16..31). Caller passes p already
// offset by hi16; loads are two contiguous 16B chunks.
__device__ __forceinline__ v16h load_frag_a(const h16* p) {
    V16 u;
    u.h[0] = *(const v8h*)(p);
    u.h[1] = *(const v8h*)(p + 8);
    return u.v;
}
// B-matrix 32x16 f16 fragment: per lane (col = lane&15), 16 contiguous K values
// (lanes 0-15: K=0..15, lanes 16-31: K=16..31 -> caller offsets by hi16).
__device__ __forceinline__ v16h load_frag_b(const h16* p) {
    V16 u;
    u.h[0] = *(const v8h*)(p);
    u.h[1] = *(const v8h*)(p + 8);
    return u.v;
}

__device__ __forceinline__ v8f wmma16(v16h a, v16h b, v8f c) {
    return __builtin_amdgcn_wmma_f32_16x16x32_f16(
        /*neg_a=*/false, a, /*neg_b=*/false, b,
        /*c_mod=*/(short)0, c, /*reuse_a=*/false, /*reuse_b=*/false);
}

// ---------------------------------------------------------------------------
// Kernel 1: row L2-normalize (512 f32 -> f16). One wave32 per row.
// Optionally also writes a dim-major transposed copy (for GEMM-2 B operand).
// ---------------------------------------------------------------------------
__global__ void __launch_bounds__(256) nrm_kernel(const float* __restrict__ src,
                                                  h16* __restrict__ dst,
                                                  h16* __restrict__ dstT,
                                                  int writeT) {
    const int lane = threadIdx.x & 31;
    const int row  = blockIdx.x * 8 + (threadIdx.x >> 5);

    const float* p = src + (size_t)row * DM + lane * 16;
    v4f q[4];
    float acc = 0.0f;
#pragma unroll
    for (int i = 0; i < 4; ++i) {
        q[i] = ((const v4f*)p)[i];
        acc += q[i].x * q[i].x + q[i].y * q[i].y + q[i].z * q[i].z + q[i].w * q[i].w;
    }
    // wave32 all-reduce
#pragma unroll
    for (int off = 16; off > 0; off >>= 1)
        acc += __shfl_xor(acc, off, 32);

    const float inv = 1.0f / fmaxf(sqrtf(acc), 1e-12f);

    union { h16 s[16]; v8h v[2]; } o;
#pragma unroll
    for (int i = 0; i < 4; ++i) {
        o.s[4 * i + 0] = (h16)(q[i].x * inv);
        o.s[4 * i + 1] = (h16)(q[i].y * inv);
        o.s[4 * i + 2] = (h16)(q[i].z * inv);
        o.s[4 * i + 3] = (h16)(q[i].w * inv);
    }
    v8h* d = (v8h*)(dst + (size_t)row * DM + lane * 16);
    d[0] = o.v[0];
    d[1] = o.v[1];

    if (writeT) {
#pragma unroll
        for (int i = 0; i < 16; ++i)
            dstT[(size_t)(lane * 16 + i) * NK + row] = o.s[i];
    }
}

// ---------------------------------------------------------------------------
// Kernel 2: fused  S = Qn@Kn^T ; P = exp(B*S-B) ; ret += P@Kn ; out = q + a*ret
// Block: 256 threads = 8 waves, one 16-query M tile. Key chunks of 128.
//   Phase A: wave w -> S subtile for keys [nb+16w, +16), full K=512 (16 wmma).
//   Phase B: wave w -> output dims [64w, 64w+64), 4 persistent C frags (16 wmma).
// P tile is double-buffered in LDS -> single s_barrier per chunk. B operands
// use a two-register-buffer rotation (both buffers live across every wmma, so
// the allocator cannot coalesce them) -> waits are loadcnt<=2, not drains.
// ---------------------------------------------------------------------------
#define PSTR 136   // LDS row stride in halves (128 keys + 8 pad, bank-skewed)

__global__ void __launch_bounds__(256) tip_fused_kernel(
        const float* __restrict__ qf,     // original queries (epilogue add)
        const h16*  __restrict__ Qn,      // (NQ x DM) normalized queries, row-major
        const h16*  __restrict__ Kn,      // (NK x DM) normalized keys, key-major
        const h16*  __restrict__ KnT,     // (DM x NK) normalized keys, dim-major
        float* __restrict__ out) {
    const int wave  = threadIdx.x >> 5;
    const int lane  = threadIdx.x & 31;
    const int lm    = lane & 15;
    const int hi8   = (lane >> 4) * 8;    // C-row offset, upper half-wave
    const int hi16  = (lane >> 4) * 16;   // B-frag K offset, upper half-wave
    const int mbase = blockIdx.x * 16;

    __shared__ h16 P[2][16 * PSTR];       // double-buffered 16x128 f16 tile

    // Resident Qn A-fragments: 16 k-steps x 8 VGPRs = 128 VGPRs, reused for
    // every one of the 1024 key subtiles this block visits.
    v16h aq[16];
    {
        const h16* qrow = Qn + (size_t)(mbase + lm) * DM + hi16;
#pragma unroll
        for (int ks = 0; ks < 16; ++ks)
            aq[ks] = load_frag_a(qrow + ks * 32);
    }

    // Persistent retrieved accumulators: 4 x 16 output dims for this wave.
    v8f cret[4];
#pragma unroll
    for (int f = 0; f < 4; ++f) cret[f] = (v8f){0, 0, 0, 0, 0, 0, 0, 0};

    for (int nb = 0; nb < NK; nb += 128) {
        h16* Pb = P[(nb >> 7) & 1];

        // ---- Phase A: S(16x16) for this wave's 16 keys, K = 512 ------------
        // Two-buffer rotation: while wmma consumes b0, b1's loads are in
        // flight; b0 is reloaded right after its use. b0 and b1 are each live
        // across the other's wmma -> distinct registers, waits loadcnt<=2.
        const h16* krow = Kn + (size_t)(nb + wave * 16 + lm) * DM + hi16;
        v8f  s  = (v8f){0, 0, 0, 0, 0, 0, 0, 0};
        v16h b0 = load_frag_b(krow);
        v16h b1 = load_frag_b(krow + 32);
#pragma unroll
        for (int ks = 0; ks < 16; ks += 2) {
            s = wmma16(aq[ks], b0, s);
            if (ks + 2 < 16) b0 = load_frag_b(krow + (ks + 2) * 32);
            s = wmma16(aq[ks + 1], b1, s);
            if (ks + 3 < 16) b1 = load_frag_b(krow + (ks + 3) * 32);
        }
        // exp(-BETA*(1-s)) = exp(BETA*s - BETA), f16 into LDS (D->A shuffle)
#pragma unroll
        for (int r = 0; r < 8; ++r) {
            float pv = __expf(fmaf(BETA_, s[r], -BETA_));
            Pb[(r + hi8) * PSTR + wave * 16 + lm] = (h16)pv;
        }

        // ---- Phase B: ret[:, 64w:64w+64] += P(16x128) @ Kn(128x64) ---------
        const h16* ktb = KnT + (size_t)(wave * 64 + lm) * NK + nb + hi16;
        v16h b2[2];
        b2[0] = load_frag_b(ktb);                     // t=0 (pre-barrier)
        b2[1] = load_frag_b(ktb + (size_t)16 * NK);   // t=1 (pre-barrier)
        __syncthreads();                              // P tile visible
        v16h a2[2];
        a2[0] = load_frag_a(&Pb[lm * PSTR + hi16]);   // kk = 0
#pragma unroll
        for (int t = 0; t < 16; ++t) {                // t = kk*4 + f
            const int kk = t >> 2, f = t & 3;
            if (f == 0 && kk < 3)                     // DS prefetch, 1 kk ahead
                a2[(kk + 1) & 1] = load_frag_a(&Pb[lm * PSTR + (kk + 1) * 32 + hi16]);
            cret[f] = wmma16(a2[kk & 1], b2[t & 1], cret[f]);
            if (t + 2 < 16) {                         // VMEM prefetch, 2 ahead
                const int tn = t + 2, kkn = tn >> 2, fn = tn & 3;
                b2[t & 1] = load_frag_b(ktb + kkn * 32 + (size_t)(fn * 16) * NK);
            }
        }
        // no second barrier: double-buffered P; writes of chunk i+2 to this
        // buffer are ordered behind sync(i+1), which follows every PhaseB(i).
    }

    // ---- Epilogue: out = q + ALPHA * retrieved  (coalesced across lanes) ---
#pragma unroll
    for (int f = 0; f < 4; ++f) {
#pragma unroll
        for (int r = 0; r < 8; ++r) {
            const int m = mbase + r + hi8;
            const int d = wave * 64 + f * 16 + lm;
            const size_t idx = (size_t)m * DM + d;
            out[idx] = qf[idx] + ALPHA_ * cret[f][r];
        }
    }
}

// ---------------------------------------------------------------------------
extern "C" void kernel_launch(void* const* d_in, const int* in_sizes, int n_in,
                              void* d_out, int out_size, void* d_ws, size_t ws_size,
                              hipStream_t stream) {
    (void)in_sizes; (void)n_in; (void)out_size; (void)ws_size;

    const float* query = (const float*)d_in[0];   // (4096, 512) f32
    const float* keys  = (const float*)d_in[1];   // (16384, 512) f32
    float* out = (float*)d_out;                   // (4096, 512) f32

    // Workspace layout (f16): Qn | Kn | KnT  -> 36 MB total
    h16* Qn  = (h16*)d_ws;
    h16* Kn  = Qn + (size_t)NQ * DM;
    h16* KnT = Kn + (size_t)NK * DM;

    nrm_kernel<<<NQ / 8, 256, 0, stream>>>(query, Qn, nullptr, 0);
    nrm_kernel<<<NK / 8, 256, 0, stream>>>(keys, Kn, KnT, 1);
    tip_fused_kernel<<<NQ / 16, 256, 0, stream>>>(query, Qn, Kn, KnT, out);
}
